// CopyGenerator_24893630448163
// MI455X (gfx1250) — matrix operand order repaired and checked by reading, not compiled, not run on hardware
//
#include <hip/hip_runtime.h>
#include <hip/hip_bf16.h>
#include <math.h>

#define PAD_IDX 1
#define BATCH   32
#define TLEN    64
#define ROWS    (BATCH * TLEN)     // 2048
#define SLEN    400
#define CVOCAB  100
#define DDIM    512
#define VOCAB   32000
#define OUTW    (VOCAB + CVOCAB)   // 32100
#define NEG_INF (-1e9f)

#define MTILE   32                 // rows per workgroup
#define NWAVES  8
#define NTILES  (VOCAB / 16)       // 2000 N-tiles of 16 columns
#define KCHUNKS (DDIM / 32)        // 16 K-chunks of 32 (bf16 WMMA K)
#define DPAD    (DDIM + 8)         // LDS row pad for A tile
#define SDT     18                 // sT row stride (floats): conflict-free

typedef __attribute__((ext_vector_type(16))) __bf16 v16bf;
typedef __attribute__((ext_vector_type(8)))  float  v8f;

// ---------------------------------------------------------------------------
// p_copy[r] = sigmoid(hidden[r,:] . Wc + bc)   -- one wave32 per row
// ---------------------------------------------------------------------------
__global__ __launch_bounds__(256)
void pcopy_kernel(const float* __restrict__ hidden,
                  const float* __restrict__ Wc,
                  const float* __restrict__ bc,
                  float* __restrict__ pc) {
  const int gw   = (blockIdx.x * blockDim.x + threadIdx.x) >> 5;
  const int lane = threadIdx.x & 31;
  if (gw >= ROWS) return;
  const float* hrow = hidden + (size_t)gw * DDIM;
  float acc = 0.0f;
  #pragma unroll 4
  for (int k = lane; k < DDIM; k += 32) acc += hrow[k] * Wc[k];
  acc += __shfl_xor(acc, 16, 32);
  acc += __shfl_xor(acc,  8, 32);
  acc += __shfl_xor(acc,  4, 32);
  acc += __shfl_xor(acc,  2, 32);
  acc += __shfl_xor(acc,  1, 32);
  if (lane == 0) pc[gw] = 1.0f / (1.0f + __expf(-(acc + bc[0])));
}

// B fragment: 32x16 bf16 tile of W at wp (already offset for K-half / column)
__device__ __forceinline__ void loadBp(const float* __restrict__ wp, v16bf& dst) {
  #pragma unroll
  for (int e = 0; e < 16; ++e)
    dst[e] = (__bf16)wp[(size_t)((e & 7) + ((e >> 3) << 4)) * VOCAB];
}

// A fragments for both 16-row subtiles; base ptrs already offset by h*8.
// ISA 16-bit A layout: e<8 -> K=kb+e ; e>=8 -> K=kb+16+(e-8)   (+h*8 in base)
__device__ __forceinline__ void loadA2(const __bf16* __restrict__ a0p,
                                       const __bf16* __restrict__ a1p,
                                       int kb, v16bf& a0, v16bf& a1) {
  #pragma unroll
  for (int e = 0; e < 16; ++e) {
    const int K = kb + (e & 7) + ((e >> 3) << 4);
    a0[e] = a0p[K];
    a1[e] = a1p[K];
  }
}

// 32x16 output tile GEMM over K=512, software-pipelined, unroll-2 ping-pong
// (no buffer-rotation movs -> no WMMA WAR hazard nops).
__device__ __forceinline__ void tile_gemm(const float* __restrict__ W,
                                          const __bf16* __restrict__ a0p,
                                          const __bf16* __restrict__ a1p,
                                          int n0, int h, int ln,
                                          v8f& acc0, v8f& acc1) {
  const float* wp = W + (size_t)(h * 8) * VOCAB + (n0 + ln);
  const size_t wstep = (size_t)32 * VOCAB;
  v16bf b0, b1, a0c, a1c, a0n, a1n;
  loadBp(wp, b0);
  loadA2(a0p, a1p, 0, a0c, a1c);
  #pragma unroll 1
  for (int kc = 0; kc < KCHUNKS; kc += 2) {
    loadBp(wp + wstep, b1);                       // chunk kc+1
    loadA2(a0p, a1p, (kc + 1) * 32, a0n, a1n);
    acc0 = __builtin_amdgcn_wmma_f32_16x16x32_bf16(false, a0c, false, b0,
                                                   (short)0, acc0, false, false);
    acc1 = __builtin_amdgcn_wmma_f32_16x16x32_bf16(false, a1c, false, b0,
                                                   (short)0, acc1, false, false);
    if (kc + 2 < KCHUNKS) {                       // prefetch chunk kc+2
      loadBp(wp + 2 * wstep, b0);
      loadA2(a0p, a1p, (kc + 2) * 32, a0c, a1c);
    }
    acc0 = __builtin_amdgcn_wmma_f32_16x16x32_bf16(false, a0n, false, b1,
                                                   (short)0, acc0, false, false);
    acc1 = __builtin_amdgcn_wmma_f32_16x16x32_bf16(false, a1n, false, b1,
                                                   (short)0, acc1, false, false);
    wp += 2 * wstep;
  }
}

// ---------------------------------------------------------------------------
// Fused generator: logits = hidden@W + b, PAD mask, flash-softmax,
// scale by (1 - p_copy), write out_prob columns [0, VOCAB).
// One workgroup = 32 rows; 8 waves stride over 2000 N-tiles of 16 cols.
// ---------------------------------------------------------------------------
__global__ __launch_bounds__(256)
void gen_softmax_kernel(const float* __restrict__ hidden,
                        const float* __restrict__ W,
                        const float* __restrict__ bvec,
                        const float* __restrict__ pc,
                        float* __restrict__ out) {
  __shared__ __bf16 sA[MTILE][DPAD];        // hidden tile, bf16
  __shared__ float  sT[NWAVES][MTILE][SDT]; // per-wave tile transpose scratch
  __shared__ float  sWm[NWAVES][MTILE];     // per-wave running max
  __shared__ float  sWs[NWAVES][MTILE];     // per-wave running sum
  __shared__ float  sM[MTILE];              // final row max
  __shared__ float  sInv[MTILE];            // (1 - p_copy) / rowsum
  __shared__ float  sPc[MTILE];

  const int tid = threadIdx.x;
  const int r0  = blockIdx.x * MTILE;

  // Stage hidden rows -> bf16 LDS (coalesced)
  for (int idx = tid; idx < MTILE * DDIM; idx += 256) {
    const int row = idx >> 9;               // /512
    const int k   = idx & (DDIM - 1);
    sA[row][k] = (__bf16)hidden[(size_t)(r0 + row) * DDIM + k];
  }
  if (tid < MTILE) sPc[tid] = pc[r0 + tid];
  __syncthreads();

  const int wid  = tid >> 5;
  const int lane = tid & 31;
  const int ln   = lane & 15;               // C/D: column within tile
  const int h    = lane >> 4;               // C/D: row-half select
  const __bf16* a0p = &sA[ln][h * 8];       // A base, rows  0..15 subtile
  const __bf16* a1p = &sA[16 + ln][h * 8];  // A base, rows 16..31 subtile

  float rm = -3.0e38f;                      // running max, row = lane
  float rs = 0.0f;                          // running exp-sum, row = lane

  // ---------------- pass 1: online row max & exp-sum ----------------
  for (int nt = wid; nt < NTILES; nt += NWAVES) {
    const int n0 = nt * 16;
    if (n0 + NWAVES * 16 < VOCAB)           // prefetch next tile of W
      __builtin_prefetch(&W[(size_t)(h * 16) * VOCAB + n0 + NWAVES * 16], 0, 1);

    v8f acc0 = {0,0,0,0,0,0,0,0};
    v8f acc1 = {0,0,0,0,0,0,0,0};
    tile_gemm(W, a0p, a1p, n0, h, ln, acc0, acc1);

    // Epilogue: transpose tile through wave-private LDS, reduce serially.
    const float bias  = bvec[n0 + ln];
    const bool  ispad = (n0 + ln) == PAD_IDX;
    float* tw = &sT[wid][0][0];
    #pragma unroll
    for (int i = 0; i < 8; ++i) {
      float v0 = acc0[i] + bias;
      float v1 = acc1[i] + bias;
      if (ispad) { v0 = NEG_INF; v1 = NEG_INF; }
      tw[(h * 8 + i) * SDT + ln]       = v0;   // rows  0..15
      tw[(16 + h * 8 + i) * SDT + ln]  = v1;   // rows 16..31
    }
    asm volatile("s_wait_dscnt 0" ::: "memory"); // wave-private, lockstep wave
    const float* trd = &sT[wid][lane][0];        // lane l reduces row l
    float vals[16];
    float tmax = -3.0e38f;
    #pragma unroll
    for (int j = 0; j < 16; ++j) {
      vals[j] = trd[j];
      tmax = fmaxf(tmax, vals[j]);
    }
    const float nm = fmaxf(rm, tmax);
    float ts = 0.0f;
    #pragma unroll
    for (int j = 0; j < 16; ++j) ts += __expf(vals[j] - nm);
    rs = rs * __expf(rm - nm) + ts;
    rm = nm;
  }

  // Combine per-wave stats (row = lane)
  sWm[wid][lane] = rm;
  sWs[wid][lane] = rs;
  __syncthreads();
  if (tid < MTILE) {
    float M = -3.0e38f;
    #pragma unroll
    for (int w = 0; w < NWAVES; ++w) M = fmaxf(M, sWm[w][tid]);
    float S = 0.0f;
    #pragma unroll
    for (int w = 0; w < NWAVES; ++w) S += sWs[w][tid] * __expf(sWm[w][tid] - M);
    sM[tid]   = M;
    sInv[tid] = (1.0f - sPc[tid]) / S;
  }
  __syncthreads();

  float Mv[2][8], Iv[2][8];
  #pragma unroll
  for (int mt = 0; mt < 2; ++mt)
    #pragma unroll
    for (int i = 0; i < 8; ++i) {
      const int row = mt * 16 + h * 8 + i;
      Mv[mt][i] = sM[row];
      Iv[mt][i] = sInv[row];
    }

  // ---------------- pass 2: recompute tiles, write probabilities ----------
  for (int nt = wid; nt < NTILES; nt += NWAVES) {
    const int n0 = nt * 16;
    v8f acc0 = {0,0,0,0,0,0,0,0};
    v8f acc1 = {0,0,0,0,0,0,0,0};
    tile_gemm(W, a0p, a1p, n0, h, ln, acc0, acc1);

    const float bias  = bvec[n0 + ln];
    const bool  ispad = (n0 + ln) == PAD_IDX;
    #pragma unroll
    for (int mt = 0; mt < 2; ++mt) {
      v8f c = mt ? acc1 : acc0;
      #pragma unroll
      for (int i = 0; i < 8; ++i) {
        float v = c[i] + bias;
        if (ispad) v = NEG_INF;
        const float p   = __expf(v - Mv[mt][i]) * Iv[mt][i];
        const int   row = r0 + mt * 16 + h * 8 + i;
        out[(size_t)row * OUTW + n0 + ln] = p;  // 64B coalesced across lanes
      }
    }
  }
}

// ---------------------------------------------------------------------------
// copy_prob[r, c] = p_copy[r] * sum_s attn[r,s] * src_map[s, r%B, c]
// written to out columns [VOCAB, VOCAB+CVOCAB)
// ---------------------------------------------------------------------------
__global__ __launch_bounds__(128)
void copyprob_kernel(const float* __restrict__ attn,
                     const float* __restrict__ srcmap,
                     const float* __restrict__ pc,
                     float* __restrict__ out) {
  const int r  = blockIdx.x;
  const int c  = threadIdx.x;
  const int bb = r & (BATCH - 1);
  if (c < CVOCAB) {
    float acc = 0.0f;
    #pragma unroll 4
    for (int s = 0; s < SLEN; ++s)
      acc += attn[(size_t)r * SLEN + s] *
             srcmap[((size_t)s * BATCH + bb) * CVOCAB + c];
    out[(size_t)r * OUTW + VOCAB + c] = pc[r] * acc;
  }
}

extern "C" void kernel_launch(void* const* d_in, const int* in_sizes, int n_in,
                              void* d_out, int out_size, void* d_ws, size_t ws_size,
                              hipStream_t stream) {
  const float* hidden = (const float*)d_in[0];
  const float* attn   = (const float*)d_in[1];
  const float* srcmap = (const float*)d_in[2];
  const float* W      = (const float*)d_in[3];
  const float* bvec   = (const float*)d_in[4];
  const float* Wc     = (const float*)d_in[5];
  const float* bc     = (const float*)d_in[6];
  float* out = (float*)d_out;
  float* pc  = (float*)d_ws;               // ROWS floats of scratch

  pcopy_kernel<<<ROWS / 8, 256, 0, stream>>>(hidden, Wc, bc, pc);
  gen_softmax_kernel<<<ROWS / MTILE, 256, 0, stream>>>(hidden, W, bvec, pc, out);
  copyprob_kernel<<<ROWS, 128, 0, stream>>>(attn, srcmap, pc, out);
}
